// AttentionalFactorizationMachine_10849087389712
// MI455X (gfx1250) — matrix operand order repaired and checked by reading, not compiled, not run on hardware
//
#include <hip/hip_runtime.h>
#include <math.h>

typedef __attribute__((ext_vector_type(16))) _Float16 v16h;
typedef __attribute__((ext_vector_type(8)))  float    v8f;

#define NFEAT  50
#define CARD   10000
#define DDIM   64
#define ADIM   64
#define NPAIR  1225      // NFEAT*(NFEAT-1)/2
#define MTILES 77        // ceil(NPAIR/16)
#define NWAVES 8
#define BLOCK  256

__global__ __launch_bounds__(BLOCK)
void afm_fused_kernel(const int*   __restrict__ inputs,   // [B, F]
                      const float* __restrict__ emb,      // [F*CARD, D]
                      const float* __restrict__ wlin,     // [F*CARD]
                      const float* __restrict__ blin,     // [1]
                      const float* __restrict__ W1,       // [D, A] row-major
                      const float* __restrict__ b1,       // [A]
                      const float* __restrict__ W2,       // [A]
                      const float* __restrict__ b2,       // [1]
                      float*       __restrict__ out)      // [B]
{
  __shared__ float          s_factor[NFEAT][DDIM + 1];   // pad -> conflict-free rows
  __shared__ _Float16       s_w1t[ADIM][DDIM + 2];       // W1 transposed, f16
  __shared__ unsigned short s_pair[NPAIR];               // (i<<8)|j
  __shared__ float          s_logit[NPAIR];
  __shared__ float          s_pooled[NPAIR];
  __shared__ float          s_red[2 * NWAVES];
  __shared__ float          s_scalar[2];                 // [0]=line_out, [1]=max logit

  const int b    = blockIdx.x;
  const int tid  = threadIdx.x;
  const int lane = tid & 31;
  const int wave = tid >> 5;
  const int half = lane >> 4;     // K-half selector (WMMA 16-bit A/B layout)
  const int nlo  = lane & 15;     // row (A) / column (B,C) within tile

  // ---------------- Phase 0: staging ----------------
  for (int p = tid; p < NPAIR; p += BLOCK) {
    int i = 0, off = 0;
    while (p >= off + (NFEAT - 1 - i)) { off += NFEAT - 1 - i; ++i; }
    int j = i + 1 + (p - off);
    s_pair[p] = (unsigned short)((i << 8) | j);
  }
  float linpart = 0.f;
  for (int f = tid; f < NFEAT; f += BLOCK) {
    int idx = inputs[b * NFEAT + f] + f * CARD;
    linpart += wlin[idx];
  }
  for (int e = tid; e < NFEAT * DDIM; e += BLOCK) {
    int f = e >> 6, d = e & 63;
    int idx = inputs[b * NFEAT + f] + f * CARD;
    s_factor[f][d] = emb[(size_t)idx * DDIM + d];
  }
  for (int e = tid; e < DDIM * ADIM; e += BLOCK) {
    int d = e >> 6, a = e & 63;
    s_w1t[a][d] = (_Float16)W1[e];
  }
  for (int m = 16; m > 0; m >>= 1) linpart += __shfl_xor(linpart, m, 32);
  if (lane == 0) s_red[wave] = linpart;
  __syncthreads();
  if (tid == 0) {
    float t = 0.f;
    for (int w = 0; w < NWAVES; ++w) t += s_red[w];
    s_scalar[0] = t + blin[0];
  }

  // ---------------- Phase 1: WMMA GEMM + fused attention head ----------------
  // B fragments (W1^T), identical for every strip: load once per wave.
  // B in contiguous k order (as A): lanes 0-15 hold K=0..15 of column nlo, lanes 16-31 hold K=16..31.
  v16h bfrag[2][4];
  #pragma unroll
  for (int kc = 0; kc < 2; ++kc) {
    #pragma unroll
    for (int nt = 0; nt < 4; ++nt) {
      int a = nt * 16 + nlo;
      int dbase = kc * 32 + half * 16;
      v16h t;
      #pragma unroll
      for (int e = 0; e < 16; ++e) t[e] = s_w1t[a][dbase + e];
      bfrag[kc][nt] = t;
    }
  }
  float b1v[4], w2v[4];
  #pragma unroll
  for (int nt = 0; nt < 4; ++nt) {
    b1v[nt] = b1[nt * 16 + nlo];
    w2v[nt] = W2[nt * 16 + nlo];
  }
  const float b2v = b2[0];

  for (int s = wave; s < MTILES; s += NWAVES) {
    int row  = s * 16 + nlo;
    int rowc = row < NPAIR ? row : NPAIR - 1;         // clamp pad rows (results discarded)
    unsigned pr = s_pair[rowc];
    const float* fi = &s_factor[pr >> 8][0];
    const float* fj = &s_factor[pr & 255][0];

    v8f acc0 = {}, acc1 = {}, acc2 = {}, acc3 = {};
    float pooled = 0.f;

    #pragma unroll
    for (int kc = 0; kc < 2; ++kc) {
      // Build A fragment in contiguous k order (k = 16*half + s, as B); fold pooled sum.
      v16h afrag;
      #pragma unroll
      for (int v = 0; v < 8; ++v) {
        int k = kc * 32 + half * 16 + 2 * v;
        float p0 = fi[k]     * fj[k];
        float p1 = fi[k + 1] * fj[k + 1];
        pooled += p0 + p1;
        afrag[2 * v]     = (_Float16)p0;
        afrag[2 * v + 1] = (_Float16)p1;
      }
      acc0 = __builtin_amdgcn_wmma_f32_16x16x32_f16(false, afrag, false, bfrag[kc][0], (short)0, acc0, false, false);
      acc1 = __builtin_amdgcn_wmma_f32_16x16x32_f16(false, afrag, false, bfrag[kc][1], (short)0, acc1, false, false);
      acc2 = __builtin_amdgcn_wmma_f32_16x16x32_f16(false, afrag, false, bfrag[kc][2], (short)0, acc2, false, false);
      acc3 = __builtin_amdgcn_wmma_f32_16x16x32_f16(false, afrag, false, bfrag[kc][3], (short)0, acc3, false, false);
    }

    // pooled[row]: lane nlo and nlo+16 each hold half the K sum of row `row`.
    pooled += __shfl_xor(pooled, 16, 32);
    if (half == 0 && row < NPAIR) s_pooled[row] = pooled;

    // logits: fold ReLU(h + b1) . W2 on the C layout (M = r + 8*half, N = nlo),
    // then reduce across the 16 lanes of each half.
    #pragma unroll
    for (int r = 0; r < 8; ++r) {
      float t = 0.f;
      t += fmaxf(acc0[r] + b1v[0], 0.f) * w2v[0];
      t += fmaxf(acc1[r] + b1v[1], 0.f) * w2v[1];
      t += fmaxf(acc2[r] + b1v[2], 0.f) * w2v[2];
      t += fmaxf(acc3[r] + b1v[3], 0.f) * w2v[3];
      #pragma unroll
      for (int m = 8; m >= 1; m >>= 1) t += __shfl_xor(t, m, 32);
      if (nlo == 0) {
        int row2 = s * 16 + half * 8 + r;
        if (row2 < NPAIR) s_logit[row2] = t + b2v;
      }
    }
  }
  __syncthreads();

  // ---------------- Phase 2: softmax over pairs + final ----------------
  float lmax = -3.402823e38f;
  for (int p = tid; p < NPAIR; p += BLOCK) lmax = fmaxf(lmax, s_logit[p]);
  for (int m = 16; m > 0; m >>= 1) lmax = fmaxf(lmax, __shfl_xor(lmax, m, 32));
  if (lane == 0) s_red[wave] = lmax;
  __syncthreads();
  if (tid == 0) {
    float t = s_red[0];
    for (int w = 1; w < NWAVES; ++w) t = fmaxf(t, s_red[w]);
    s_scalar[1] = t;
  }
  __syncthreads();
  const float gmax = s_scalar[1];

  float esum = 0.f, wsum = 0.f;
  for (int p = tid; p < NPAIR; p += BLOCK) {
    float e = __expf(s_logit[p] - gmax);
    esum += e;
    wsum += e * s_pooled[p];
  }
  for (int m = 16; m > 0; m >>= 1) {
    esum += __shfl_xor(esum, m, 32);
    wsum += __shfl_xor(wsum, m, 32);
  }
  if (lane == 0) { s_red[wave] = esum; s_red[NWAVES + wave] = wsum; }
  __syncthreads();
  if (tid == 0) {
    float es = 0.f, ws = 0.f;
    for (int w = 0; w < NWAVES; ++w) { es += s_red[w]; ws += s_red[NWAVES + w]; }
    out[b] = s_scalar[0] + ws / es;
  }
}

extern "C" void kernel_launch(void* const* d_in, const int* in_sizes, int n_in,
                              void* d_out, int out_size, void* d_ws, size_t ws_size,
                              hipStream_t stream) {
  const int*   inputs = (const int*)d_in[0];
  const float* emb    = (const float*)d_in[1];
  const float* wlin   = (const float*)d_in[2];
  const float* blin   = (const float*)d_in[3];
  const float* W1     = (const float*)d_in[4];
  const float* b1     = (const float*)d_in[5];
  const float* W2     = (const float*)d_in[6];
  const float* b2     = (const float*)d_in[7];
  float* out = (float*)d_out;

  const int batch = in_sizes[0] / NFEAT;   // inputs is [B, F]
  afm_fused_kernel<<<batch, BLOCK, 0, stream>>>(inputs, emb, wlin, blin,
                                                W1, b1, W2, b2, out);
  (void)n_in; (void)out_size; (void)d_ws; (void)ws_size;
}
